// HGTLayer_13013750907158
// MI455X (gfx1250) — compile-verified, matrix operand order checked by reading, not run on hardware
//
#include <hip/hip_runtime.h>

// ---------------- types ----------------
typedef __attribute__((ext_vector_type(16))) __bf16 bf16x16;
typedef __attribute__((ext_vector_type(8)))  __bf16 bf16x8;
typedef __attribute__((ext_vector_type(8)))  float  v8f;

// order-preserving float <-> uint encoding for atomic max
__device__ __forceinline__ unsigned f2ord(float f) {
  unsigned b = __float_as_uint(f);
  return (b & 0x80000000u) ? ~b : (b | 0x80000000u);
}
__device__ __forceinline__ float ord2f(unsigned u) {
  unsigned b = (u & 0x80000000u) ? (u & 0x7FFFFFFFu) : ~u;
  return __uint_as_float(b);
}

// ---------------- conversion kernels ----------------
__global__ void hgt_cvt_bf16(const float* __restrict__ in, __bf16* __restrict__ out, int n) {
  int t = blockIdx.x * blockDim.x + threadIdx.x;
  if (t < n) out[t] = (__bf16)in[t];
}

__global__ void hgt_cvt_scale_bf16(const float* __restrict__ in, __bf16* __restrict__ out,
                                   float scale, int n) {
  int t = blockIdx.x * blockDim.x + threadIdx.x;
  if (t < n) out[t] = (__bf16)(in[t] * scale);
}

__global__ void hgt_init_u32(unsigned* __restrict__ p, unsigned val, int n) {
  int t = blockIdx.x * blockDim.x + threadIdx.x;
  if (t < n) p[t] = val;
}

// ---------------- weight folding ----------------
// WT[n][k] = sum_d w[k][h*32+d] * relm[h][d][e]   (n = h*32+e), stored bf16 transposed (n-major)
// bias_out[n] = sum_d b[h*32+d] * relm[h][d][e]
__global__ void hgt_fold_kv(const float* __restrict__ w, const float* __restrict__ b,
                            const float* __restrict__ relm,
                            __bf16* __restrict__ WT, float* __restrict__ bias_out) {
  int t = blockIdx.x * blockDim.x + threadIdx.x;   // 65536 threads exactly
  int n = t >> 8, k = t & 255;
  int h = n >> 5, e = n & 31;
  const float* rm = relm + h * 1024;               // [32][32]
  const float* wr = w + k * 256 + h * 32;
  float s = 0.f;
#pragma unroll 8
  for (int d = 0; d < 32; ++d) s += wr[d] * rm[d * 32 + e];
  WT[n * 256 + k] = (__bf16)s;
  if (k == 0) {
    const float* br = b + h * 32;
    float sb = 0.f;
#pragma unroll 8
    for (int d = 0; d < 32; ++d) sb += br[d] * rm[d * 32 + e];
    bias_out[n] = sb;
  }
}

// WT[n][k] = w[k][n]  (plain transpose + bf16)
__global__ void hgt_fold_tr(const float* __restrict__ w, __bf16* __restrict__ WT) {
  int t = blockIdx.x * blockDim.x + threadIdx.x;   // 65536 threads exactly
  int n = t >> 8, k = t & 255;
  WT[n * 256 + k] = (__bf16)w[k * 256 + n];
}

// ---------------- WMMA GEMM: C[M,256] = A[M,256](bf16) * WT^T + bias ----------------
// WT stored n-major [256 out][256 in] so B fragments load contiguously.
// Fragment layout per CDNA5 ISA 7.12.2 (16-bit A 16x32): lane&15 = row,
// lanes 0-15 hold K {0..7,16..23}, lanes 16-31 hold K {8..15,24..31}.
__device__ __forceinline__ bf16x16 hgt_load_frag(const __bf16* __restrict__ rowptr,
                                                 int kb, int lane_hi) {
  const bf16x8* p = (const bf16x8*)(rowptr + kb + lane_hi);
  bf16x8 lo = p[0];
  bf16x8 hi = p[2];  // +16 elements
  return __builtin_shufflevector(lo, hi, 0, 1, 2, 3, 4, 5, 6, 7,
                                 8, 9, 10, 11, 12, 13, 14, 15);
}

__global__ void hgt_gemm_bf16(const __bf16* __restrict__ A, const __bf16* __restrict__ WT,
                              const float* __restrict__ bias, float* __restrict__ C) {
  int lane = threadIdx.x & 31;
  int wave = threadIdx.x >> 5;          // 0..3, each wave owns 64 output cols
  int m0 = blockIdx.x * 16;
  int n0 = wave * 64;
  int lane_lo = lane & 15;
  int lane_hi = (lane & 16) ? 8 : 0;
  const __bf16* arow = A + (size_t)(m0 + lane_lo) * 256;
  const __bf16* brow = WT + (size_t)(n0 + lane_lo) * 256;

  v8f acc[4] = {};
#pragma unroll
  for (int kb = 0; kb < 256; kb += 32) {
    bf16x16 af = hgt_load_frag(arow, kb, lane_hi);
#pragma unroll
    for (int j = 0; j < 4; ++j) {
      bf16x16 bfrag = hgt_load_frag(brow + (size_t)j * 16 * 256, kb, lane_hi);
      acc[j] = __builtin_amdgcn_wmma_f32_16x16x32_bf16(
          false, af, false, bfrag, (short)0, acc[j], false, false);
    }
  }

  int mbase = m0 + ((lane & 16) ? 8 : 0);
#pragma unroll
  for (int j = 0; j < 4; ++j) {
    int n = n0 + j * 16 + lane_lo;
    float bv = bias[n];
#pragma unroll
    for (int r = 0; r < 8; ++r)
      C[(size_t)(mbase + r) * 256 + n] = acc[j][r] + bv;
  }
}

// ---------------- edge kernels ----------------
__global__ void hgt_edge_score(const int* __restrict__ src, const int* __restrict__ dst,
                               const float* __restrict__ K, const float* __restrict__ Q,
                               const float* __restrict__ pri,
                               float* __restrict__ score, unsigned* __restrict__ mx, int E) {
  int t = blockIdx.x * blockDim.x + threadIdx.x;
  if (t >= E * 8) return;
  int e = t >> 3, h = t & 7;
  int s = src[e], d = dst[e];
  const float* qp = Q + (size_t)d * 256 + h * 32;
  const float* kp = K + (size_t)s * 256 + h * 32;
  float acc = 0.f;
#pragma unroll 8
  for (int i = 0; i < 32; ++i) acc += qp[i] * kp[i];
  acc *= pri[h] * 0.17677669529663687f;  // 1/sqrt(32)
  score[t] = acc;
  atomicMax(mx + (size_t)d * 8 + h, f2ord(acc));
}

__global__ void hgt_edge_exp(const int* __restrict__ dst, float* __restrict__ score,
                             const unsigned* __restrict__ mx, float* __restrict__ den, int E) {
  int t = blockIdx.x * blockDim.x + threadIdx.x;
  if (t >= E * 8) return;
  int e = t >> 3, h = t & 7;
  int d = dst[e];
  float m = ord2f(mx[(size_t)d * 8 + h]);
  float ex = expf(score[t] - m);
  score[t] = ex;  // reuse buffer: now holds exp
  atomicAdd(den + (size_t)d * 8 + h, ex);
}

__global__ void hgt_edge_agg(const int* __restrict__ src, const int* __restrict__ dst,
                             const float* __restrict__ score, const float* __restrict__ den,
                             const float* __restrict__ V, float* __restrict__ mu, int E) {
  int e = blockIdx.x;
  int tid = threadIdx.x;  // 0..255 covers all H*DK components
  int h = tid >> 5;
  int s = src[e], d = dst[e];
  float attn = score[(size_t)e * 8 + h] / den[(size_t)d * 8 + h];
  atomicAdd(mu + (size_t)d * 256 + tid, V[(size_t)s * 256 + tid] * attn);
}

// ---------------- skip + layernorm output ----------------
__global__ void hgt_node_out(const float* __restrict__ Clin, const float* __restrict__ t,
                             const float* __restrict__ skip, int nt,
                             const float* __restrict__ lns, const float* __restrict__ lnb,
                             float* __restrict__ out) {
  int row = blockIdx.x, tid = threadIdx.x;
  float alpha = 1.f / (1.f + expf(-skip[nt]));
  size_t idx = (size_t)row * 256 + tid;
  float o = Clin[idx] * alpha + (1.f - alpha) * t[idx];

  __shared__ float red[8];
  int lane = tid & 31, wid = tid >> 5;

  float v = o;
#pragma unroll
  for (int off = 16; off; off >>= 1) v += __shfl_xor(v, off, 32);
  if (lane == 0) red[wid] = v;
  __syncthreads();
  float sum = 0.f;
#pragma unroll
  for (int i = 0; i < 8; ++i) sum += red[i];
  float mean = sum * (1.f / 256.f);
  float dv = o - mean;
  __syncthreads();

  v = dv * dv;
#pragma unroll
  for (int off = 16; off; off >>= 1) v += __shfl_xor(v, off, 32);
  if (lane == 0) red[wid] = v;
  __syncthreads();
  float vs = 0.f;
#pragma unroll
  for (int i = 0; i < 8; ++i) vs += red[i];
  float var = vs * (1.f / 256.f);

  out[idx] = dv * rsqrtf(var + 1e-5f) * lns[tid] + lnb[tid];
}

// ---------------- host launch ----------------
extern "C" void kernel_launch(void* const* d_in, const int* in_sizes, int n_in,
                              void* d_out, int out_size, void* d_ws, size_t ws_size,
                              hipStream_t stream) {
  (void)n_in; (void)out_size; (void)ws_size;
  const float* h_paper  = (const float*)d_in[0];
  const float* h_author = (const float*)d_in[1];
  const float* t_paper  = (const float*)d_in[2];
  const float* t_author = (const float*)d_in[3];
  const float* k_w = (const float*)d_in[4];
  const float* k_b = (const float*)d_in[5];
  const float* q_w = (const float*)d_in[6];
  const float* q_b = (const float*)d_in[7];
  const float* v_w = (const float*)d_in[8];
  const float* v_b = (const float*)d_in[9];
  const float* a_w = (const float*)d_in[10];
  const float* a_b = (const float*)d_in[11];
  const float* rel_pri = (const float*)d_in[12];
  const float* rel_att = (const float*)d_in[13];
  const float* rel_msg = (const float*)d_in[14];
  const float* skip     = (const float*)d_in[15];
  const float* ln_scale = (const float*)d_in[16];
  const float* ln_bias  = (const float*)d_in[17];
  const int* srcs[3] = {(const int*)d_in[18], (const int*)d_in[20], (const int*)d_in[22]};
  const int* dsts[3] = {(const int*)d_in[19], (const int*)d_in[21], (const int*)d_in[23]};

  const int NP = in_sizes[0] / 256;   // 20000
  const int NA = in_sizes[1] / 256;   // 10000
  const int E  = in_sizes[18];        // 100000
  float* out = (float*)d_out;

  // workspace carve-up (256B aligned)
  char* base = (char*)d_ws;
  size_t off = 0;
  auto alloc = [&](size_t bytes) -> void* {
    void* p = base + off;
    off += bytes;
    off = (off + 255) & ~(size_t)255;
    return p;
  };
  __bf16* hpb = (__bf16*)alloc((size_t)NP * 256 * 2);
  __bf16* hab = (__bf16*)alloc((size_t)NA * 256 * 2);
  __bf16* tpb = (__bf16*)alloc((size_t)NP * 256 * 2);
  __bf16* tab = (__bf16*)alloc((size_t)NA * 256 * 2);
  float* Qp = (float*)alloc((size_t)NP * 256 * 4);
  float* Qa = (float*)alloc((size_t)NA * 256 * 4);
  int Nmax = NP > NA ? NP : NA;
  float* Kbuf = (float*)alloc((size_t)Nmax * 256 * 4);
  float* Vbuf = (float*)alloc((size_t)Nmax * 256 * 4);
  float* score = (float*)alloc((size_t)E * 8 * 4);
  unsigned* mx = (unsigned*)alloc((size_t)Nmax * 8 * 4);
  float* den = (float*)alloc((size_t)Nmax * 8 * 4);
  float* mu_p = (float*)alloc((size_t)NP * 256 * 4);
  float* mu_a = (float*)alloc((size_t)NA * 256 * 4);
  __bf16* mu_bf = (__bf16*)alloc((size_t)(NP + NA) * 256 * 2);
  float* Clin = (float*)alloc((size_t)(NP + NA) * 256 * 4);
  __bf16* KWT = (__bf16*)alloc((size_t)3 * 65536 * 2);
  __bf16* VWT = (__bf16*)alloc((size_t)3 * 65536 * 2);
  __bf16* QWT = (__bf16*)alloc((size_t)2 * 65536 * 2);
  __bf16* AWT = (__bf16*)alloc((size_t)2 * 65536 * 2);
  float* kbf = (float*)alloc((size_t)3 * 256 * 4);
  float* vbf = (float*)alloc((size_t)3 * 256 * 4);

  // 1) convert features to bf16
  hgt_cvt_bf16<<<(NP * 256 + 255) / 256, 256, 0, stream>>>(h_paper, hpb, NP * 256);
  hgt_cvt_bf16<<<(NA * 256 + 255) / 256, 256, 0, stream>>>(h_author, hab, NA * 256);
  hgt_cvt_bf16<<<(NP * 256 + 255) / 256, 256, 0, stream>>>(t_paper, tpb, NP * 256);
  hgt_cvt_bf16<<<(NA * 256 + 255) / 256, 256, 0, stream>>>(t_author, tab, NA * 256);

  // 2) fold relation transforms into combined transposed bf16 weights
  const int nts[3] = {1, 0, 0};  // src ntype per etype
  for (int r = 0; r < 3; ++r) {
    hgt_fold_kv<<<256, 256, 0, stream>>>(k_w + nts[r] * 65536, k_b + nts[r] * 256,
                                         rel_att + r * 8192, KWT + r * 65536, kbf + r * 256);
    hgt_fold_kv<<<256, 256, 0, stream>>>(v_w + nts[r] * 65536, v_b + nts[r] * 256,
                                         rel_msg + r * 8192, VWT + r * 65536, vbf + r * 256);
  }
  for (int nt = 0; nt < 2; ++nt) {
    hgt_fold_tr<<<256, 256, 0, stream>>>(q_w + nt * 65536, QWT + nt * 65536);
    hgt_fold_tr<<<256, 256, 0, stream>>>(a_w + nt * 65536, AWT + nt * 65536);
  }

  // 3) Q projections (per dst ntype)
  hgt_gemm_bf16<<<NP / 16, 128, 0, stream>>>(tpb, QWT, q_b, Qp);
  hgt_gemm_bf16<<<NA / 16, 128, 0, stream>>>(tab, QWT + 65536, q_b + 256, Qa);

  hipMemsetAsync(mu_p, 0, (size_t)NP * 256 * 4, stream);
  hipMemsetAsync(mu_a, 0, (size_t)NA * 256 * 4, stream);

  // 4) per-etype: K/V GEMMs, edge softmax, aggregation
  struct EtDesc { const __bf16* hb; int Nsrc; const float* Q; int Ndst; float* mu; };
  EtDesc et[3] = {
      {hab, NA, Qp, NP, mu_p},   // author -> paper (rel 0)
      {hpb, NP, Qp, NP, mu_p},   // paper  -> paper (rel 1)
      {hpb, NP, Qa, NA, mu_a},   // paper  -> author (rel 2)
  };
  for (int r = 0; r < 3; ++r) {
    hgt_gemm_bf16<<<et[r].Nsrc / 16, 128, 0, stream>>>(et[r].hb, KWT + r * 65536,
                                                       kbf + r * 256, Kbuf);
    hgt_gemm_bf16<<<et[r].Nsrc / 16, 128, 0, stream>>>(et[r].hb, VWT + r * 65536,
                                                       vbf + r * 256, Vbuf);
    int nd8 = et[r].Ndst * 8;
    hgt_init_u32<<<(nd8 + 255) / 256, 256, 0, stream>>>(mx, 0x007FFFFFu, nd8);  // enc(-inf)
    hipMemsetAsync(den, 0, (size_t)nd8 * 4, stream);
    int e8 = E * 8;
    hgt_edge_score<<<(e8 + 255) / 256, 256, 0, stream>>>(srcs[r], dsts[r], Kbuf, et[r].Q,
                                                         rel_pri + r * 8, score, mx, E);
    hgt_edge_exp<<<(e8 + 255) / 256, 256, 0, stream>>>(dsts[r], score, mx, den, E);
    hgt_edge_agg<<<E, 256, 0, stream>>>(srcs[r], dsts[r], score, den, Vbuf, et[r].mu, E);
  }

  // 5) cross-reducer mean for paper, then output projection + skip + LN
  hgt_cvt_scale_bf16<<<(NP * 256 + 255) / 256, 256, 0, stream>>>(mu_p, mu_bf, 0.5f, NP * 256);
  hgt_cvt_scale_bf16<<<(NA * 256 + 255) / 256, 256, 0, stream>>>(mu_a, mu_bf + (size_t)NP * 256,
                                                                 1.0f, NA * 256);
  hgt_gemm_bf16<<<NP / 16, 128, 0, stream>>>(mu_bf, AWT, a_b, Clin);
  hgt_gemm_bf16<<<NA / 16, 128, 0, stream>>>(mu_bf + (size_t)NP * 256, AWT + 65536,
                                             a_b + 256, Clin + (size_t)NP * 256);

  hgt_node_out<<<NP, 256, 0, stream>>>(Clin, t_paper, skip, 0, ln_scale, ln_bias, out);
  hgt_node_out<<<NA, 256, 0, stream>>>(Clin + (size_t)NP * 256, t_author, skip, 1,
                                       ln_scale + 256, ln_bias + 256, out + (size_t)NP * 256);
}
